// Wavenet_24945170055752
// MI455X (gfx1250) — matrix operand order repaired and checked by reading, not compiled, not run on hardware
//
#include <hip/hip_runtime.h>
#include <hip/hip_bf16.h>
#include <stdint.h>

typedef unsigned short u16;
typedef __attribute__((ext_vector_type(16))) __bf16 v16bf;
typedef __attribute__((ext_vector_type(8)))  float  v8f;

#define N_BATCH 16
#define L_SEQ   8192
#define PAD     128                 // max (K-1)*dil = 2*64
#define XROWS   (L_SEQ + PAD)
#define XSTR    ((size_t)XROWS * 32)
#define N_LAYERS 14
#define TILES_PER_BATCH (L_SEQ / 16)   // 512

// ---------------- scalar helpers ----------------
__device__ __forceinline__ u16 f2bf(float f) {
  unsigned u = __builtin_bit_cast(unsigned, f);
  unsigned r = (u + 0x7FFFu + ((u >> 16) & 1u)) >> 16;
  return (u16)r;
}
__device__ __forceinline__ float bf2f(u16 s) {
  return __builtin_bit_cast(float, (unsigned)s << 16);
}
__device__ __forceinline__ float fsigmoid(float x) { return 1.0f / (1.0f + __expf(-x)); }
__device__ __forceinline__ float ftanh(float x) {
  float e = __expf(-2.0f * x);
  return (1.0f - e) / (1.0f + e);
}

// ---------------- WMMA operand loaders (wave32) ----------------
// A tile 16x32 bf16, row-major src with row stride S (elems), K offset k0.
// Lane r=lane&15 holds row r; lanes<16: K={k0..k0+7, k0+16..k0+23}; lanes>=16: +8.
__device__ __forceinline__ v16bf ld_A(const u16* base, int S, int k0) {
  int lane = threadIdx.x & 31, r = lane & 15, hi = lane >> 4;
  const u16* p = base + r * S + k0 + hi * 8;
  union { uint4 q[2]; v16bf v; } u;
  u.q[0] = *(const uint4*)(p);
  u.q[1] = *(const uint4*)(p + 16);
  return u.v;
}
// B tile 32x16 bf16 from position-major activations: row stride S elems per
// position, K offset k0. Lane n=lane&15 = column; K={k0+hi*8..+7, k0+16+hi*8..+7}.
__device__ __forceinline__ v16bf ld_B(const u16* base, int S, int k0) {
  int lane = threadIdx.x & 31, c = lane & 15, hi = lane >> 4;
  const u16* p = base + c * S + k0 + hi * 8;
  union { uint4 q[2]; v16bf v; } u;
  u.q[0] = *(const uint4*)(p);
  u.q[1] = *(const uint4*)(p + 16);
  return u.v;
}
// C/D bias init: p points at row-tile base (16 f32); lane takes hi*8..+7.
__device__ __forceinline__ v8f ld_c_f32(const float* p) {
  int hi = (threadIdx.x >> 4) & 1;
  const float* q = p + hi * 8;
  v8f r;
#pragma unroll
  for (int j = 0; j < 8; ++j) r[j] = q[j];
  return r;
}
__device__ __forceinline__ uint4 pack_bf8(v8f v) {
  union { u16 s[8]; uint4 q; } u;
#pragma unroll
  for (int j = 0; j < 8; ++j) u.s[j] = f2bf(v[j]);
  return u.q;
}
__device__ __forceinline__ v8f unpack_bf8(uint4 q) {
  union { u16 s[8]; uint4 q; } u; u.q = q;
  v8f r;
#pragma unroll
  for (int j = 0; j < 8; ++j) r[j] = bf2f(u.s[j]);
  return r;
}
__device__ __forceinline__ v8f wmma_bf(v16bf a, v16bf b, v8f c) {
  return __builtin_amdgcn_wmma_f32_16x16x32_bf16(false, a, false, b, (short)0, c,
                                                 false, false);
}

// ---------------- weight repack to bf16 (WMMA-A layouts) ----------------
__global__ void k_prepw(const float* fw, const float* gw, const float* cfw_i,
                        const float* cgw_i, const float* rw, const float* sw,
                        const float* fb, const float* gb, const float* cfb,
                        const float* cgb, const float* rb, const float* sb,
                        const float* w1, const float* w2,
                        u16* wft, u16* wgt, u16* cfw, u16* cgw, u16* resw,
                        u16* skw, u16* w1b, u16* w2b,
                        float* fbc, float* gbc, float* rbc, float* skb) {
  int tid = blockIdx.x * blockDim.x + threadIdx.x;
  int nt = gridDim.x * blockDim.x;
  // filter/gate: [14][32][32][3] -> [14][3][32][32]
  for (int i = tid; i < N_LAYERS * 3 * 1024; i += nt) {
    int l = i / 3072, r = i % 3072, k = r / 1024, oc = r % 1024;
    int o = oc >> 5, c = oc & 31;
    int src = ((l * 32 + o) * 32 + c) * 3 + k;
    wft[i] = f2bf(fw[src]);
    wgt[i] = f2bf(gw[src]);
  }
  // cond proj: already [O][C] row-major, just convert
  for (int i = tid; i < N_LAYERS * 32 * 128; i += nt) {
    cfw[i] = f2bf(cfw_i[i]);
    cgw[i] = f2bf(cgw_i[i]);
  }
  for (int i = tid; i < N_LAYERS * 1024; i += nt) resw[i] = f2bf(rw[i]);
  // skip: [14][256][32] -> fused [256][14*32]
  for (int i = tid; i < N_LAYERS * 256 * 32; i += nt) {
    int l = i / 8192, o = (i % 8192) >> 5, c = i & 31;
    skw[o * (N_LAYERS * 32) + l * 32 + c] = f2bf(sw[i]);
  }
  // input stack weights (already row-major [O][C])
  for (int i = tid; i < 32 * 256; i += nt) w1b[i] = f2bf(w1[i]);
  for (int i = tid; i < 32 * 32; i += nt)  w2b[i] = f2bf(w2[i]);
  for (int i = tid; i < N_LAYERS * 32; i += nt) {
    fbc[i] = fb[i] + cfb[i];
    gbc[i] = gb[i] + cgb[i];
    rbc[i] = rb[i];
  }
  for (int i = tid; i < 256; i += nt) {
    float s = 0.f;
    for (int l = 0; l < N_LAYERS; ++l) s += sb[l * 256 + i];
    skb[i] = s;
  }
}

// ---- LDS-tiled transpose: [C][L] f32 -> [L][C] bf16 (per batch) ----------
__global__ void __launch_bounds__(256) k_transpose(const float* __restrict__ in,
                                                   u16* __restrict__ outb,
                                                   int C, int cpb) {
  __shared__ float tile[64][65];
  int b = blockIdx.x;
  int n = b / ((L_SEQ / 64) * cpb);
  int rem = b % ((L_SEQ / 64) * cpb);
  int t0 = (rem / cpb) * 64;
  int c0 = (rem % cpb) * 64;
  const float* ip = in + (size_t)n * C * L_SEQ;
  u16* op = outb + (size_t)n * L_SEQ * C;
#pragma unroll
  for (int i = 0; i < 16; ++i) {
    int idx = threadIdx.x + i * 256;
    int c = idx >> 6, t = idx & 63;           // lanes consecutive along t: coalesced
    tile[t][c] = ip[(size_t)(c0 + c) * L_SEQ + t0 + t];
  }
  __syncthreads();
#pragma unroll
  for (int i = 0; i < 16; ++i) {
    int idx = threadIdx.x + i * 256;
    int t = idx >> 6, c = idx & 63;           // lanes consecutive along c: coalesced
    op[(size_t)(t0 + t) * C + c0 + c] = f2bf(tile[t][c]);
  }
}

// ---- zero the causal pads of both ping-pong x buffers --------------------
__global__ void k_zeropad(u16* __restrict__ x0, u16* __restrict__ x1) {
  int idx = blockIdx.x * blockDim.x + threadIdx.x;  // 16*PAD*32 = 65536
  int n = idx / (PAD * 32), r = idx % (PAD * 32);
  x0[(size_t)n * XSTR + r] = 0;
  x1[(size_t)n * XSTR + r] = 0;
}

// ---- input stack as WMMA: 256->32 relu -> 32->32 relu --------------------
__global__ void __launch_bounds__(128) k_in(const u16* __restrict__ inb,
                                            const u16* __restrict__ w1b,
                                            const float* __restrict__ b1,
                                            const u16* __restrict__ w2b,
                                            const float* __restrict__ b2,
                                            u16* __restrict__ x0) {
  __shared__ __align__(16) u16 lh[4][16 * 32];
  int wv = (blockIdx.x << 2) | (threadIdx.x >> 5);
  int n = wv >> 9;
  int t0 = (wv & (TILES_PER_BATCH - 1)) << 4;
  int lane = threadIdx.x & 31, col = lane & 15, hi = lane >> 4;

  v8f h0 = ld_c_f32(b1), h1 = ld_c_f32(b1 + 16);
  const u16* ibase = inb + ((size_t)n * L_SEQ + t0) * 256;
#pragma unroll
  for (int kk = 0; kk < 8; ++kk) {
    v16bf B = ld_B(ibase, 256, kk * 32);
    h0 = wmma_bf(ld_A(w1b, 256, kk * 32), B, h0);
    h1 = wmma_bf(ld_A(w1b + 16 * 256, 256, kk * 32), B, h1);
  }
#pragma unroll
  for (int j = 0; j < 8; ++j) { h0[j] = fmaxf(h0[j], 0.f); h1[j] = fmaxf(h1[j], 0.f); }
  u16* myh = lh[threadIdx.x >> 5];
  *(uint4*)(myh + col * 32 + hi * 8) = pack_bf8(h0);
  *(uint4*)(myh + col * 32 + 16 + hi * 8) = pack_bf8(h1);
  asm volatile("" ::: "memory");  // keep DS store->load order (same-wave, in-order DS)
  v16bf Bh = ld_B(myh, 32, 0);
  v8f y0 = ld_c_f32(b2), y1 = ld_c_f32(b2 + 16);
  y0 = wmma_bf(ld_A(w2b, 32, 0), Bh, y0);
  y1 = wmma_bf(ld_A(w2b + 512, 32, 0), Bh, y1);
#pragma unroll
  for (int j = 0; j < 8; ++j) { y0[j] = fmaxf(y0[j], 0.f); y1[j] = fmaxf(y1[j], 0.f); }
  u16* xp = x0 + (size_t)n * XSTR + (size_t)(PAD + t0 + col) * 32 + hi * 8;
  *(uint4*)(xp) = pack_bf8(y0);
  *(uint4*)(xp + 16) = pack_bf8(y1);
}

// ---------------- one WaveNet layer (dilated conv + gate + residual) ------
__global__ void __launch_bounds__(128) k_layer(
    const u16* __restrict__ xin, u16* __restrict__ xout, u16* __restrict__ zall,
    const u16* __restrict__ condb, const u16* __restrict__ wft,
    const u16* __restrict__ wgt, const u16* __restrict__ cfw,
    const u16* __restrict__ cgw, const u16* __restrict__ resw,
    const float* __restrict__ fbc, const float* __restrict__ gbc,
    const float* __restrict__ rbc, int l, int dil) {
  __shared__ __align__(16) u16 lz[4][16 * 32];
  int wv = (blockIdx.x << 2) | (threadIdx.x >> 5);
  int n = wv >> 9;
  int t0 = (wv & (TILES_PER_BATCH - 1)) << 4;
  int lane = threadIdx.x & 31, col = lane & 15, hi = lane >> 4;

  const u16* wfL = wft + l * 3072;
  const u16* wgL = wgt + l * 3072;
  const u16* cfL = cfw + l * 4096;
  const u16* cgL = cgw + l * 4096;
  const u16* wrL = resw + l * 1024;
  const float* fbL = fbc + l * 32;
  const float* gbL = gbc + l * 32;
  const float* rbL = rbc + l * 32;

  v8f f0 = ld_c_f32(fbL), f1 = ld_c_f32(fbL + 16);
  v8f g0 = ld_c_f32(gbL), g1 = ld_c_f32(gbL + 16);

  const u16* xb = xin + (size_t)n * XSTR + (size_t)(PAD + t0) * 32;
  // 3 dilated causal taps, K = 32 channels each
#pragma unroll
  for (int k = 0; k < 3; ++k) {
    v16bf B = ld_B(xb - (size_t)(2 - k) * dil * 32, 32, 0);
    f0 = wmma_bf(ld_A(wfL + k * 1024, 32, 0), B, f0);
    f1 = wmma_bf(ld_A(wfL + k * 1024 + 512, 32, 0), B, f1);
    g0 = wmma_bf(ld_A(wgL + k * 1024, 32, 0), B, g0);
    g1 = wmma_bf(ld_A(wgL + k * 1024 + 512, 32, 0), B, g1);
  }
  // condition projections, K = 128
  const u16* cbase = condb + ((size_t)n * L_SEQ + t0) * 128;
#pragma unroll
  for (int kk = 0; kk < 4; ++kk) {
    v16bf Bc = ld_B(cbase, 128, kk * 32);
    f0 = wmma_bf(ld_A(cfL, 128, kk * 32), Bc, f0);
    f1 = wmma_bf(ld_A(cfL + 16 * 128, 128, kk * 32), Bc, f1);
    g0 = wmma_bf(ld_A(cgL, 128, kk * 32), Bc, g0);
    g1 = wmma_bf(ld_A(cgL + 16 * 128, 128, kk * 32), Bc, g1);
  }
  // z = tanh(f) * sigmoid(g)
  v8f z0, z1;
#pragma unroll
  for (int j = 0; j < 8; ++j) {
    z0[j] = ftanh(f0[j]) * fsigmoid(g0[j]);
    z1[j] = ftanh(f1[j]) * fsigmoid(g1[j]);
  }
  uint4 qz0 = pack_bf8(z0), qz1 = pack_bf8(z1);
  // store z: global (for the fused skip GEMM) + LDS (C-layout -> B-layout)
  u16* zp = zall + ((size_t)(l * N_BATCH + n) * L_SEQ + t0 + col) * 32 + hi * 8;
  *(uint4*)(zp) = qz0;
  *(uint4*)(zp + 16) = qz1;
  u16* myz = lz[threadIdx.x >> 5];
  *(uint4*)(myz + col * 32 + hi * 8) = qz0;
  *(uint4*)(myz + col * 32 + 16 + hi * 8) = qz1;
  asm volatile("" ::: "memory");  // keep DS store->load order (same-wave, in-order DS)
  // residual: x_out = res_w @ z + res_b + x_in
  const u16* xinp = xin + (size_t)n * XSTR + (size_t)(PAD + t0 + col) * 32 + hi * 8;
  v8f r0 = ld_c_f32(rbL) + unpack_bf8(*(const uint4*)(xinp));
  v8f r1 = ld_c_f32(rbL + 16) + unpack_bf8(*(const uint4*)(xinp + 16));
  v16bf Bz = ld_B(myz, 32, 0);
  r0 = wmma_bf(ld_A(wrL, 32, 0), Bz, r0);
  r1 = wmma_bf(ld_A(wrL + 512, 32, 0), Bz, r1);
  u16* xop = xout + (size_t)n * XSTR + (size_t)(PAD + t0 + col) * 32 + hi * 8;
  *(uint4*)(xop) = pack_bf8(r0);
  *(uint4*)(xop + 16) = pack_bf8(r1);
}

// ------------- fused skip GEMM (K = 14*32 = 448) + output head -----------
__global__ void __launch_bounds__(128) k_final(
    const u16* __restrict__ zall, const u16* __restrict__ skw,
    const float* __restrict__ skb, const float* __restrict__ o1w,
    const float* __restrict__ o1b, const float* __restrict__ o2w,
    const float* __restrict__ o2b, float* __restrict__ out) {
  int wv = (blockIdx.x << 2) | (threadIdx.x >> 5);
  int n = wv >> 9;
  int t0 = (wv & (TILES_PER_BATCH - 1)) << 4;
  int lane = threadIdx.x & 31, col = lane & 15, hi = lane >> 4;

  v8f acc[16];
#pragma unroll
  for (int rt = 0; rt < 16; ++rt) acc[rt] = ld_c_f32(skb + rt * 16);
  for (int l = 0; l < N_LAYERS; ++l) {
    const u16* zb = zall + ((size_t)(l * N_BATCH + n) * L_SEQ + t0) * 32;
    if (l + 1 < N_LAYERS)  // gfx1250 global_prefetch of next layer's z tile
      __builtin_prefetch(zall + ((size_t)((l + 1) * N_BATCH + n) * L_SEQ + t0) * 32, 0, 0);
    v16bf B = ld_B(zb, 32, 0);
#pragma unroll
    for (int rt = 0; rt < 16; ++rt)
      acc[rt] = wmma_bf(ld_A(skw + rt * 16 * (N_LAYERS * 32), N_LAYERS * 32, l * 32),
                        B, acc[rt]);
  }
  // out1 (256 -> 2): per-lane partial dot over its 128 channels
  float p0 = 0.f, p1 = 0.f;
#pragma unroll
  for (int rt = 0; rt < 16; ++rt) {
#pragma unroll
    for (int j = 0; j < 8; ++j) {
      int c = rt * 16 + hi * 8 + j;
      float s = acc[rt][j];
      p0 = fmaf(o1w[c], s, p0);
      p1 = fmaf(o1w[256 + c], s, p1);
    }
  }
  p0 += __shfl_xor(p0, 16, 32);
  p1 += __shfl_xor(p1, 16, 32);
  if (hi == 0) {
    float a0 = fmaxf(p0 + o1b[0], 0.f);
    float a1 = fmaxf(p1 + o1b[1], 0.f);
    float b0 = fmaxf(fmaf(o2w[0], a0, fmaf(o2w[1], a1, o2b[0])), 0.f);
    float b1 = fmaxf(fmaf(o2w[2], a0, fmaf(o2w[3], a1, o2b[1])), 0.f);
    int t = t0 + col;
    out[((size_t)n * 2 + 0) * L_SEQ + t] = b0;
    out[((size_t)n * 2 + 1) * L_SEQ + t] = b1;
  }
}

extern "C" void kernel_launch(void* const* d_in, const int* in_sizes, int n_in,
                              void* d_out, int out_size, void* d_ws, size_t ws_size,
                              hipStream_t stream) {
  (void)in_sizes; (void)n_in; (void)out_size; (void)ws_size;
  const float* input    = (const float*)d_in[0];
  const float* cond     = (const float*)d_in[1];
  const float* in1_w    = (const float*)d_in[2];
  const float* in1_b    = (const float*)d_in[3];
  const float* in2_w    = (const float*)d_in[4];
  const float* in2_b    = (const float*)d_in[5];
  const float* filter_w = (const float*)d_in[6];
  const float* filter_b = (const float*)d_in[7];
  const float* gate_w   = (const float*)d_in[8];
  const float* gate_b   = (const float*)d_in[9];
  const float* cf_w     = (const float*)d_in[10];
  const float* cf_b     = (const float*)d_in[11];
  const float* cg_w     = (const float*)d_in[12];
  const float* cg_b     = (const float*)d_in[13];
  const float* res_w    = (const float*)d_in[14];
  const float* res_b    = (const float*)d_in[15];
  const float* skip_w   = (const float*)d_in[16];
  const float* skip_b   = (const float*)d_in[17];
  const float* out1_w   = (const float*)d_in[18];
  const float* out1_b   = (const float*)d_in[19];
  const float* out2_w   = (const float*)d_in[20];
  const float* out2_b   = (const float*)d_in[21];

  char* ws = (char*)d_ws;
  size_t off = 0;
  auto alloc = [&](size_t bytes) -> char* {
    char* p = ws + off;
    off += (bytes + 255) & ~(size_t)255;
    return p;
  };
  u16* inb   = (u16*)alloc((size_t)N_BATCH * L_SEQ * 256 * 2);
  u16* condb = (u16*)alloc((size_t)N_BATCH * L_SEQ * 128 * 2);
  u16* x0    = (u16*)alloc((size_t)N_BATCH * XROWS * 32 * 2);
  u16* x1    = (u16*)alloc((size_t)N_BATCH * XROWS * 32 * 2);
  u16* zall  = (u16*)alloc((size_t)N_LAYERS * N_BATCH * L_SEQ * 32 * 2);
  u16* wft   = (u16*)alloc((size_t)N_LAYERS * 3 * 1024 * 2);
  u16* wgt   = (u16*)alloc((size_t)N_LAYERS * 3 * 1024 * 2);
  u16* cfw   = (u16*)alloc((size_t)N_LAYERS * 32 * 128 * 2);
  u16* cgw   = (u16*)alloc((size_t)N_LAYERS * 32 * 128 * 2);
  u16* resw  = (u16*)alloc((size_t)N_LAYERS * 1024 * 2);
  u16* skw   = (u16*)alloc((size_t)256 * N_LAYERS * 32 * 2);
  u16* w1b   = (u16*)alloc((size_t)32 * 256 * 2);
  u16* w2b   = (u16*)alloc((size_t)32 * 32 * 2);
  float* fbc = (float*)alloc((size_t)N_LAYERS * 32 * 4);
  float* gbc = (float*)alloc((size_t)N_LAYERS * 32 * 4);
  float* rbc = (float*)alloc((size_t)N_LAYERS * 32 * 4);
  float* skb = (float*)alloc(256 * 4);

  k_prepw<<<64, 256, 0, stream>>>(filter_w, gate_w, cf_w, cg_w, res_w, skip_w,
                                  filter_b, gate_b, cf_b, cg_b, res_b, skip_b,
                                  in1_w, in2_w,
                                  wft, wgt, cfw, cgw, resw, skw, w1b, w2b,
                                  fbc, gbc, rbc, skb);
  // transpose input [n][256][L] -> bf16 [n][L][256], cond -> [n][L][128]
  k_transpose<<<N_BATCH * (L_SEQ / 64) * 4, 256, 0, stream>>>(input, inb, 256, 4);
  k_transpose<<<N_BATCH * (L_SEQ / 64) * 2, 256, 0, stream>>>(cond, condb, 128, 2);
  k_zeropad<<<(N_BATCH * PAD * 32) / 256, 256, 0, stream>>>(x0, x1);

  const int nblocks = (N_BATCH * TILES_PER_BATCH) / 4;  // 4 waves / block
  k_in<<<nblocks, 128, 0, stream>>>(inb, w1b, in1_b, w2b, in2_b, x0);

  u16* xin = x0; u16* xout = x1;
  for (int l = 0; l < N_LAYERS; ++l) {
    int dil = 1 << (l % 7);
    k_layer<<<nblocks, 128, 0, stream>>>(xin, xout, zall, condb, wft, wgt, cfw,
                                         cgw, resw, fbc, gbc, rbc, l, dil);
    u16* tmp = xin; xin = xout; xout = tmp;
  }
  k_final<<<nblocks, 128, 0, stream>>>(zall, skw, skb, out1_w, out1_b, out2_w,
                                       out2_b, (float*)d_out);
}
